// FrameFiberConv_50019189129837
// MI455X (gfx1250) — matrix operand order repaired and checked by reading, not compiled, not run on hardware
//
#include <hip/hip_runtime.h>

// ---------- CDNA5 WMMA types ----------
typedef __bf16 bf16_t;
typedef bf16_t bf16x16 __attribute__((ext_vector_type(16)));
typedef float  floatx8 __attribute__((ext_vector_type(8)));

// f32 -> bf16 round-to-nearest-even (bit trick; inputs ~N(0,1), no inf/nan care)
__device__ __forceinline__ unsigned short f32_to_bf16_rne(float f) {
  unsigned int u = __builtin_bit_cast(unsigned int, f);
  u += 0x7FFFu + ((u >> 16) & 1u);
  return (unsigned short)(u >> 16);
}
__device__ __forceinline__ unsigned int pack_bf16x2(float lo, float hi) {
  return (unsigned int)f32_to_bf16_rne(lo) | ((unsigned int)f32_to_bf16_rne(hi) << 16);
}
__device__ __forceinline__ bf16_t to_bf16(float f) {
  unsigned short s = f32_to_bf16_rne(f);
  union { unsigned short s; bf16_t h; } u;
  u.s = s;
  return u.h;
}

// One wave processes one vertex per grid-stride iteration:
//   out[f] = relu(max_d( sum_{w,c} x[(d+w)%16, c] * K[w,c,f] ) + bias[f])
// as 8x V_WMMA_F32_16X16X32_BF16 (K=256), circulant A built via LDS addressing.
__global__ void __launch_bounds__(256)
FrameFiberConv_kernel(const float* __restrict__ x,     // [nVert, 16, 16]
                      const float* __restrict__ kern,  // [16, 16, 16] (w,c,f)
                      const float* __restrict__ bias,  // [16]
                      float* __restrict__ out,         // [nVert, 16]
                      int nVert) {
  // Per-wave bf16 tile: 16 rows x 16 cols x 2B = 512B (row = 8 dwords)
  __shared__ __align__(16) unsigned int lds[8][16][8];

  const int tid  = threadIdx.x;
  const int wave = tid >> 5;
  const int lane = tid & 31;
  const int f    = lane & 15;   // N column (and A row base M)
  const int half = lane >> 4;   // selects K sub-range per the 16-bit A/B layouts

  // ---- Resident B fragments (kernel reused by every vertex) ----
  // B (32x16 bf16): lane<16 -> N=f, K=0..15 ; lane>=16 -> N=f, K=16..31.
  // Chunk kk: K<16 -> (w=2kk, c=K) ; K>=16 -> (w=2kk+1, c=K-16)  => w = 2kk+half.
  bf16x16 Bfrag[8];
#pragma unroll
  for (int kk = 0; kk < 8; ++kk) {
    const int w = 2 * kk + half;
    const float* kb = kern + w * 256 + f;  // K[w][c][f], stride 16 over c
    bf16x16 b;
#pragma unroll
    for (int c = 0; c < 16; ++c) b[c] = to_bf16(kb[c * 16]);
    Bfrag[kk] = b;
  }
  const float bias_f = bias[f];

  int v = blockIdx.x * 8 + wave;
  const int stride = gridDim.x * 8;

  // ---- Software pipeline: global loads for vertex v already in flight ----
  float4 q0 = {0.f, 0.f, 0.f, 0.f}, q1 = {0.f, 0.f, 0.f, 0.f};
  if (v < nVert) {
    const float4* p = (const float4*)(x + (size_t)v * 256);
    q0 = p[lane * 2 + 0];
    q1 = p[lane * 2 + 1];
  }

  while (v < nVert) {
    const int vn = v + stride;

    // ---- Stage current vertex tile to LDS as bf16, row-major [16][16] ----
    uint4 st;
    st.x = pack_bf16x2(q0.x, q0.y);
    st.y = pack_bf16x2(q0.z, q0.w);
    st.z = pack_bf16x2(q1.x, q1.y);
    st.w = pack_bf16x2(q1.z, q1.w);
    *(uint4*)&lds[wave][lane >> 1][(lane & 1) * 4] = st;
    __builtin_amdgcn_wave_barrier();  // DS in-order per wave; block reordering

    // Far prefetch (fills L2, no LOADcnt): distance 8 iterations covers HBM
    // latency. Near prefetch at distance 2 pulls toward the WGP caches.
    if (v + 8 * stride < nVert)
      __builtin_prefetch(x + (size_t)(v + 8 * stride) * 256 + lane * 8, 0, 1);
    if (v + 2 * stride < nVert)
      __builtin_prefetch(x + (size_t)(v + 2 * stride) * 256 + lane * 8, 0, 3);

    // Issue NEXT vertex's real loads now: their latency overlaps the
    // ds-gather + 8 WMMAs below. (Wave-uniform branch; EXEC full at WMMAs.)
    float4 n0 = q0, n1 = q1;
    if (vn < nVert) {
      const float4* p = (const float4*)(x + (size_t)vn * 256);
      n0 = p[lane * 2 + 0];
      n1 = p[lane * 2 + 1];
    }
    __builtin_amdgcn_sched_barrier(0);  // pin global loads above the gather

    // ---- Gather ALL A fragments (16x ds_load_b128) BEFORE the WMMA chain.
    // A (16x32 bf16): lane M = lane&15; row (M+2kk+half)%16,
    // VGPR0-3 = c 0..7, VGPR4-7 = c 8..15.
    uint4 Aq[16];
#pragma unroll
    for (int kk = 0; kk < 8; ++kk) {
      const int r0 = (f + 2 * kk + half) & 15;
      const int r1 = r0;
      Aq[2 * kk + 0] = *(const uint4*)&lds[wave][r0][0];
      Aq[2 * kk + 1] = *(const uint4*)&lds[wave][r1][4];
    }
    // Hard fence: scheduler must NOT sink the 16 ds_loads into the WMMA
    // chain (waitcnt pass then emits staged s_wait_dscnt, one LDS latency
    // per vertex instead of eight).
    __builtin_amdgcn_sched_barrier(0);

    floatx8 acc = {0.f, 0.f, 0.f, 0.f, 0.f, 0.f, 0.f, 0.f};
#pragma unroll
    for (int kk = 0; kk < 8; ++kk) {
      union { bf16x16 v; uint4 q[2]; } A;
      A.q[0] = Aq[2 * kk + 0];
      A.q[1] = Aq[2 * kk + 1];
      acc = __builtin_amdgcn_wmma_f32_16x16x32_bf16(
          /*neg_a=*/false, A.v, /*neg_b=*/false, Bfrag[kk],
          /*c_mod=*/(short)0, acc, /*reuse_a=*/false, /*reuse_b=*/false);
    }

    // ---- Epilogue: max over d, then bias + relu (relu/max commute) ----
    float m = acc[0];
#pragma unroll
    for (int r = 1; r < 8; ++r) m = fmaxf(m, acc[r]);
    m = fmaxf(m, __shfl_xor(m, 16, 32));   // merge the two M-halves (wave32)
    m = fmaxf(m + bias_f, 0.0f);
    if (half == 0) out[(size_t)v * 16 + f] = m;

    q0 = n0;
    q1 = n1;
    v  = vn;
  }
}

extern "C" void kernel_launch(void* const* d_in, const int* in_sizes, int n_in,
                              void* d_out, int out_size, void* d_ws, size_t ws_size,
                              hipStream_t stream) {
  const float* x    = (const float*)d_in[0];
  const float* kern = (const float*)d_in[1];
  const float* bias = (const float*)d_in[2];
  float* out        = (float*)d_out;

  const int nVert = in_sizes[0] / 256;  // B*V vertices, 16x16 floats each
  // Long-lived waves (grid-stride) so the distance-8 prefetch window matures:
  // 1024 blocks * 8 waves = 8192 waves -> ~24 vertices per wave at nVert=200k.
  int blocks = (nVert + 7) / 8;
  if (blocks > 1024) blocks = 1024;
  if (blocks < 1) blocks = 1;

  FrameFiberConv_kernel<<<blocks, 256, 0, stream>>>(x, kern, bias, out, nVert);
}
